// Experts_50405736186174
// MI455X (gfx1250) — matrix operand order repaired and checked by reading, not compiled, not run on hardware
//
#include <hip/hip_runtime.h>

// ---------------------------------------------------------------------------
// MoE BERT encoder layer for MI455X (gfx1250, wave32, WMMA bf16).
// ---------------------------------------------------------------------------

#define H 768
#define NH 12
#define HD 64
#define FF 3072
#define E 4
#define B 64
#define S 128

typedef __attribute__((ext_vector_type(16))) __bf16 v16bf;
typedef __attribute__((ext_vector_type(8)))  __bf16 v8bf;
typedef __attribute__((ext_vector_type(8)))  float  v8f;

__device__ __forceinline__ v8f wmma_bf16(v16bf a, v16bf b, v8f c) {
    // D(16x16,f32) = A(16x32,bf16) x B(32x16,bf16) + C
    return __builtin_amdgcn_wmma_f32_16x16x32_bf16(
        /*neg_a=*/false, a, /*neg_b=*/false, b,
        /*c_mod=*/(short)0, c, /*reuse_a=*/false, /*reuse_b=*/false);
}

__device__ __forceinline__ float gelu_tanh(float x) {
    float x3 = x * x * x;
    return 0.5f * x * (1.0f + tanhf(0.7978845608028654f * (x + 0.044715f * x3)));
}

// ---------------------------------------------------------------------------
// Routing: eid[b] = argmin_e || mean_s(X[b,s,:]) - centers[e,:] ||^2
// ---------------------------------------------------------------------------
__global__ __launch_bounds__(256)
void route_kernel(const float* __restrict__ X, const float* __restrict__ C,
                  int* __restrict__ eid) {
    int b = blockIdx.x, tid = threadIdx.x;
    float pd[E] = {0.f, 0.f, 0.f, 0.f};
    for (int h = tid; h < H; h += 256) {
        const float* col = X + (size_t)b * S * H + h;
        float s = 0.f;
        for (int si = 0; si < S; ++si) s += col[(size_t)si * H];
        float m = s * (1.0f / (float)S);
        #pragma unroll
        for (int e = 0; e < E; ++e) {
            float d = m - C[e * H + h];
            pd[e] += d * d;
        }
    }
    __shared__ float red[E][256];
    #pragma unroll
    for (int e = 0; e < E; ++e) red[e][tid] = pd[e];
    __syncthreads();
    for (int off = 128; off >= 1; off >>= 1) {
        if (tid < off) {
            #pragma unroll
            for (int e = 0; e < E; ++e) red[e][tid] += red[e][tid + off];
        }
        __syncthreads();
    }
    if (tid == 0) {
        int best = 0; float bv = red[0][0];
        #pragma unroll
        for (int e = 1; e < E; ++e)
            if (red[e][0] < bv) { bv = red[e][0]; best = e; }
        eid[b] = best;
    }
}

// ---------------------------------------------------------------------------
// Generic per-sample expert GEMM:  Y[b] = act(X[b] @ W[eid[b]] + bias) (+ Res)
// Block tile 128x128, 8 waves (4x2), wave tile 32x64 (2x4 WMMA frags).
// fp32 -> bf16 conversion in registers, packed v8bf (ds_store_b128) staging,
// double-buffered LDS so next tile's loads overlap current tile's WMMAs.
// ---------------------------------------------------------------------------
__device__ __forceinline__ void stage_tiles(const float* __restrict__ Xb,
                                            const float* __restrict__ W,
                                            int m0, int n0, int K, int N, int k0,
                                            __bf16 (*Xs)[40], __bf16 (*Ws)[40],
                                            int tid) {
    // X tile: 128 rows x 32 k. Thread owns 16 contiguous k of one row.
    {
        int m = tid & 127, ko = (tid >> 7) * 16;
        const float* src = Xb + (size_t)(m0 + m) * K + k0 + ko;
        float xv[16];
        #pragma unroll
        for (int j = 0; j < 16; ++j) xv[j] = src[j];
        v8bf p0, p1;
        #pragma unroll
        for (int j = 0; j < 8; ++j) { p0[j] = (__bf16)xv[j]; p1[j] = (__bf16)xv[8 + j]; }
        *(v8bf*)&Xs[m][ko]     = p0;
        *(v8bf*)&Xs[m][ko + 8] = p1;
    }
    // W tile transposed: Ws[n][k]. Thread owns 16 k of one n; for fixed j the
    // 32 lanes of a wave read 32 consecutive n -> coalesced global dwords.
    {
        int n = tid & 127, ko = (tid >> 7) * 16;
        float wv[16];
        #pragma unroll
        for (int j = 0; j < 16; ++j)
            wv[j] = W[(size_t)(k0 + ko + j) * N + n0 + n];
        v8bf p0, p1;
        #pragma unroll
        for (int j = 0; j < 8; ++j) { p0[j] = (__bf16)wv[j]; p1[j] = (__bf16)wv[8 + j]; }
        *(v8bf*)&Ws[n][ko]     = p0;
        *(v8bf*)&Ws[n][ko + 8] = p1;
    }
}

__global__ __launch_bounds__(256)
void gemm_kernel(const float* __restrict__ X,    // [B][M][K]
                 const float* __restrict__ Wst,  // [E][K][N]
                 const float* __restrict__ Bst,  // [E][N]
                 const int*   __restrict__ eid,
                 const float* __restrict__ Res,  // [B][M][N] or nullptr
                 float* __restrict__ Y,          // [B][M][N]
                 int M, int K, int N, int act) {
    int b = blockIdx.z;
    int e = eid[b];
    const float* Xb   = X   + (size_t)b * M * K;
    const float* W    = Wst + (size_t)e * K * N;
    const float* bias = Bst + (size_t)e * N;
    const float* Rb   = Res ? (Res + (size_t)b * M * N) : nullptr;
    float*       Yb   = Y   + (size_t)b * M * N;

    int n0 = blockIdx.x * 128, m0 = blockIdx.y * 128;
    int tid = threadIdx.x, lane = tid & 31, w = tid >> 5;
    int wr = w >> 1, wc = w & 1;           // 4x2 wave grid
    int half = lane >> 4, l16 = lane & 15;

    __shared__ __bf16 Xs[2][128][40];      // [buf][m][k], 80B row stride (16B-mult)
    __shared__ __bf16 Ws[2][128][40];      // [buf][n][k] (transposed)

    v8f acc[2][4];
    #pragma unroll
    for (int i = 0; i < 2; ++i)
        #pragma unroll
        for (int j = 0; j < 4; ++j)
            #pragma unroll
            for (int r = 0; r < 8; ++r) acc[i][j][r] = 0.0f;

    stage_tiles(Xb, W, m0, n0, K, N, 0, Xs[0], Ws[0], tid);
    __syncthreads();

    int nk = K >> 5;
    for (int t = 0; t < nk; ++t) {
        int cur = t & 1;
        // Speculative prefetch two tiles ahead.
        if (t + 2 < nk) {
            __builtin_prefetch(&Xb[(size_t)(m0 + (tid & 127)) * K + (t + 2) * 32], 0, 1);
            __builtin_prefetch(&W[(size_t)((t + 2) * 32 + (tid >> 5)) * N + n0 + (tid & 31) * 4], 0, 1);
        }
        // Stage next tile into the other buffer while computing this one.
        if (t + 1 < nk)
            stage_tiles(Xb, W, m0, n0, K, N, (t + 1) * 32, Xs[cur ^ 1], Ws[cur ^ 1], tid);

        v16bf a[2], bb[4];
        #pragma unroll
        for (int mt = 0; mt < 2; ++mt) {
            int m = wr * 32 + mt * 16 + l16;
            #pragma unroll
            for (int j = 0; j < 8; ++j) {
                a[mt][j]     = Xs[cur][m][half * 8 + j];       // K {0..7}/{8..15}
                a[mt][8 + j] = Xs[cur][m][16 + half * 8 + j];  // K {16..23}/{24..31}
            }
        }
        #pragma unroll
        for (int nt = 0; nt < 4; ++nt) {
            int n = wc * 64 + nt * 16 + l16;
            #pragma unroll
            for (int j = 0; j < 16; ++j)
                bb[nt][j] = Ws[cur][n][half * 8 + (j & 7) + (j >> 3) * 16];  // K {0..7,16..23}/{8..15,24..31}
        }
        #pragma unroll
        for (int mt = 0; mt < 2; ++mt)
            #pragma unroll
            for (int nt = 0; nt < 4; ++nt)
                acc[mt][nt] = wmma_bf16(a[mt], bb[nt], acc[mt][nt]);
        __syncthreads();
    }

    // Epilogue: bias (+residual) (+gelu). C layout: VGPR r -> M = r + 8*half, N = l16.
    #pragma unroll
    for (int mt = 0; mt < 2; ++mt)
        #pragma unroll
        for (int nt = 0; nt < 4; ++nt) {
            int n = n0 + wc * 64 + nt * 16 + l16;
            float bv = bias[n];
            #pragma unroll
            for (int r = 0; r < 8; ++r) {
                int m = m0 + wr * 32 + mt * 16 + half * 8 + r;
                float v = acc[mt][nt][r] + bv;
                if (Rb) v += Rb[(size_t)m * N + n];
                if (act == 1) v = gelu_tanh(v);
                Yb[(size_t)m * N + n] = v;
            }
        }
}

// ---------------------------------------------------------------------------
// Attention: one block per (b, head). 4 waves, each owns 32 query rows.
// scores = (Q K^T) * 0.125 + mask -> softmax (register shuffles) -> P @ V.
// LDS region R1 aliases K-tile (stage1) then P-tile (stage2); Vs transposed.
// ---------------------------------------------------------------------------
__global__ __launch_bounds__(128)
void attn_kernel(const float* __restrict__ Q, const float* __restrict__ Kt,
                 const float* __restrict__ V, const float* __restrict__ mask,
                 float* __restrict__ CTX) {
    int bh = blockIdx.x;
    int b = bh / NH, h = bh % NH;
    const float* Qb = Q    + (size_t)b * S * H + h * HD;
    const float* Kb = Kt   + (size_t)b * S * H + h * HD;
    const float* Vb = V    + (size_t)b * S * H + h * HD;
    const float* mb = mask + (size_t)b * S;
    float*       Ob = CTX  + (size_t)b * S * H + h * HD;

    int tid = threadIdx.x, lane = tid & 31, w = tid >> 5;
    int half = lane >> 4, l16 = lane & 15;

    // R1: stage1 = K rows [128][72] bf16; stage2 = P rows [128][136] bf16.
    __shared__ __bf16 R1[S * 136];      // 34,816 B
    __shared__ __bf16 Vs[HD * 136];     // 17,408 B (transposed V: [d][key])

    // --- stage 1: K tile, one thread per row, packed v8bf stores ---
    {
        int r = tid;
        const float* src = Kb + (size_t)r * H;
        #pragma unroll
        for (int sg = 0; sg < 8; ++sg) {
            v8bf p;
            #pragma unroll
            for (int j = 0; j < 8; ++j) p[j] = (__bf16)src[sg * 8 + j];
            *(v8bf*)&R1[r * 72 + sg * 8] = p;
        }
    }
    // Q fragments straight from global (wave rows = w*32 .. w*32+31).
    v16bf qf[2][2];
    #pragma unroll
    for (int mt = 0; mt < 2; ++mt)
        #pragma unroll
        for (int kb = 0; kb < 2; ++kb) {
            int m = w * 32 + mt * 16 + l16;
            const float* qr = Qb + (size_t)m * H + kb * 32;
            #pragma unroll
            for (int j = 0; j < 8; ++j) {
                qf[mt][kb][j]     = (__bf16)qr[half * 8 + j];
                qf[mt][kb][8 + j] = (__bf16)qr[16 + half * 8 + j];
            }
        }
    float mval[8];
    #pragma unroll
    for (int nt = 0; nt < 8; ++nt) mval[nt] = mb[nt * 16 + l16];
    __syncthreads();

    // --- scores: 32 rows x 128 keys per wave, K-dim = HD = 64 (2 steps) ---
    v8f sc[2][8];
    #pragma unroll
    for (int mt = 0; mt < 2; ++mt)
        #pragma unroll
        for (int nt = 0; nt < 8; ++nt)
            #pragma unroll
            for (int r = 0; r < 8; ++r) sc[mt][nt][r] = 0.0f;

    #pragma unroll
    for (int nt = 0; nt < 8; ++nt) {
        v16bf kf[2];
        int n = nt * 16 + l16;  // key index (column of K^T)
        #pragma unroll
        for (int kb = 0; kb < 2; ++kb)
            #pragma unroll
            for (int j = 0; j < 16; ++j)
                kf[kb][j] = R1[n * 72 + kb * 32 + half * 8 + (j & 7) + (j >> 3) * 16];
        #pragma unroll
        for (int mt = 0; mt < 2; ++mt) {
            sc[mt][nt] = wmma_bf16(qf[mt][0], kf[0], sc[mt][nt]);
            sc[mt][nt] = wmma_bf16(qf[mt][1], kf[1], sc[mt][nt]);
        }
    }

    // --- softmax in registers: each score row lives in one 16-lane half ---
    #pragma unroll
    for (int mt = 0; mt < 2; ++mt)
        #pragma unroll
        for (int r = 0; r < 8; ++r) {
            float vals[8];
            float mx = -3.0e38f;
            #pragma unroll
            for (int nt = 0; nt < 8; ++nt) {
                vals[nt] = sc[mt][nt][r] * 0.125f + mval[nt];
                mx = fmaxf(mx, vals[nt]);
            }
            #pragma unroll
            for (int msk = 8; msk >= 1; msk >>= 1)
                mx = fmaxf(mx, __shfl_xor(mx, msk));
            float sum = 0.f;
            #pragma unroll
            for (int nt = 0; nt < 8; ++nt) {
                vals[nt] = __expf(vals[nt] - mx);
                sum += vals[nt];
            }
            #pragma unroll
            for (int msk = 8; msk >= 1; msk >>= 1)
                sum += __shfl_xor(sum, msk);
            float inv = 1.0f / sum;
            #pragma unroll
            for (int nt = 0; nt < 8; ++nt) sc[mt][nt][r] = vals[nt] * inv;
        }

    __syncthreads();  // all waves done reading K region before P overwrites it

    // --- store P (bf16) into R1; stage V transposed into Vs (packed) ---
    #pragma unroll
    for (int mt = 0; mt < 2; ++mt)
        #pragma unroll
        for (int nt = 0; nt < 8; ++nt)
            #pragma unroll
            for (int r = 0; r < 8; ++r) {
                int m = w * 32 + mt * 16 + half * 8 + r;
                int c = nt * 16 + l16;
                R1[m * 136 + c] = (__bf16)sc[mt][nt][r];
            }
    {
        int d = tid >> 1, keyoff = (tid & 1) * 64;
        #pragma unroll
        for (int sg = 0; sg < 8; ++sg) {
            v8bf p;
            #pragma unroll
            for (int j = 0; j < 8; ++j)
                p[j] = (__bf16)Vb[(size_t)(keyoff + sg * 8 + j) * H + d];
            *(v8bf*)&Vs[d * 136 + keyoff + sg * 8] = p;
        }
    }
    __syncthreads();

    // --- ctx = P @ V: 32 rows x 64 cols per wave, K-dim = 128 (4 steps) ---
    v8f o[2][4];
    #pragma unroll
    for (int mt = 0; mt < 2; ++mt)
        #pragma unroll
        for (int nt = 0; nt < 4; ++nt)
            #pragma unroll
            for (int r = 0; r < 8; ++r) o[mt][nt][r] = 0.0f;

    #pragma unroll
    for (int kb = 0; kb < 4; ++kb) {
        v16bf pa[2];
        #pragma unroll
        for (int mt = 0; mt < 2; ++mt) {
            int m = w * 32 + mt * 16 + l16;
            int base = m * 136 + kb * 32;
            #pragma unroll
            for (int j = 0; j < 8; ++j) {
                pa[mt][j]     = R1[base + half * 8 + j];
                pa[mt][8 + j] = R1[base + 16 + half * 8 + j];
            }
        }
        v16bf vbf[4];
        #pragma unroll
        for (int nt = 0; nt < 4; ++nt) {
            int n = nt * 16 + l16;  // output hd column
            #pragma unroll
            for (int j = 0; j < 16; ++j)
                vbf[nt][j] = Vs[n * 136 + kb * 32 + half * 8 + (j & 7) + (j >> 3) * 16];
        }
        #pragma unroll
        for (int mt = 0; mt < 2; ++mt)
            #pragma unroll
            for (int nt = 0; nt < 4; ++nt)
                o[mt][nt] = wmma_bf16(pa[mt], vbf[nt], o[mt][nt]);
    }

    #pragma unroll
    for (int mt = 0; mt < 2; ++mt)
        #pragma unroll
        for (int nt = 0; nt < 4; ++nt)
            #pragma unroll
            for (int r = 0; r < 8; ++r) {
                int m = w * 32 + mt * 16 + half * 8 + r;
                int c = nt * 16 + l16;
                Ob[(size_t)m * H + c] = o[mt][nt][r];
            }
}

// ---------------------------------------------------------------------------
// LayerNorm over last dim (768), per-sample expert gamma/beta.
// One block per (s, b) row.
// ---------------------------------------------------------------------------
__global__ __launch_bounds__(256)
void ln_kernel(const float* __restrict__ T, const float* __restrict__ Gst,
               const float* __restrict__ Bst, const int* __restrict__ eid,
               float* __restrict__ O) {
    int s = blockIdx.x, b = blockIdx.y;
    int e = eid[b];
    const float* row = T + ((size_t)b * S + s) * H;
    float*       orow = O + ((size_t)b * S + s) * H;
    const float* g  = Gst + (size_t)e * H;
    const float* be = Bst + (size_t)e * H;
    int tid = threadIdx.x;

    float x0 = row[tid], x1 = row[tid + 256], x2 = row[tid + 512];
    float p = x0 + x1 + x2;
    __shared__ float red[8];
    #pragma unroll
    for (int m = 16; m >= 1; m >>= 1) p += __shfl_xor(p, m);
    if ((tid & 31) == 0) red[tid >> 5] = p;
    __syncthreads();
    float mean = (red[0] + red[1] + red[2] + red[3] +
                  red[4] + red[5] + red[6] + red[7]) * (1.0f / (float)H);
    float d0 = x0 - mean, d1 = x1 - mean, d2 = x2 - mean;
    float pv = d0 * d0 + d1 * d1 + d2 * d2;
    __syncthreads();  // everyone done reading red before reuse
    #pragma unroll
    for (int m = 16; m >= 1; m >>= 1) pv += __shfl_xor(pv, m);
    if ((tid & 31) == 0) red[tid >> 5] = pv;
    __syncthreads();
    float var = (red[0] + red[1] + red[2] + red[3] +
                 red[4] + red[5] + red[6] + red[7]) * (1.0f / (float)H);
    float rstd = rsqrtf(var + 1e-12f);
    orow[tid]       = d0 * rstd * g[tid]       + be[tid];
    orow[tid + 256] = d1 * rstd * g[tid + 256] + be[tid + 256];
    orow[tid + 512] = d2 * rstd * g[tid + 512] + be[tid + 512];
}

// ---------------------------------------------------------------------------
// Host-side orchestration.
// ---------------------------------------------------------------------------
extern "C" void kernel_launch(void* const* d_in, const int* in_sizes, int n_in,
                              void* d_out, int out_size, void* d_ws, size_t ws_size,
                              hipStream_t stream) {
    const float* hs      = (const float*)d_in[0];   // [64,128,768]
    const float* amask   = (const float*)d_in[1];   // [64,128]
    const float* centers = (const float*)d_in[2];   // [4,768]
    const float* wq = (const float*)d_in[3];
    const float* wk = (const float*)d_in[4];
    const float* wv = (const float*)d_in[5];
    const float* wo = (const float*)d_in[6];
    const float* bq = (const float*)d_in[7];
    const float* bk = (const float*)d_in[8];
    const float* bv = (const float*)d_in[9];
    const float* bo = (const float*)d_in[10];
    const float* ln1_g = (const float*)d_in[11];
    const float* ln1_b = (const float*)d_in[12];
    const float* w1 = (const float*)d_in[13];
    const float* b1 = (const float*)d_in[14];
    const float* w2 = (const float*)d_in[15];
    const float* b2 = (const float*)d_in[16];
    const float* ln2_g = (const float*)d_in[17];
    const float* ln2_b = (const float*)d_in[18];

    const size_t SZ_ACT = (size_t)B * S * H;   // 6,291,456 f32
    const size_t SZ_FF  = (size_t)B * S * FF;  // 25,165,824 f32

    char* base = (char*)d_ws;
    int*   eid = (int*)base;                    // 64 ints (256 B slot)
    float* Q   = (float*)(base + 256);
    float* Kt  = Q   + SZ_ACT;
    float* Vt  = Kt  + SZ_ACT;
    float* CTX = Vt  + SZ_ACT;
    float* F1  = CTX + SZ_ACT;
    // Reuse dead buffers for later stages:
    float* T1 = Q;    // pre-LN1 (Q dead after attention)
    float* A1 = Kt;   // LN1 output (K dead after attention)
    float* T2 = Vt;   // pre-LN2 (V dead after attention)
    float* OUT = (float*)d_out;

    // 1) routing
    route_kernel<<<B, 256, 0, stream>>>(hs, centers, eid);

    // 2) Q/K/V projections (bias fused); block tile 128x128
    dim3 gP(H / 128, S / 128, B);
    gemm_kernel<<<gP, 256, 0, stream>>>(hs, wq, bq, eid, nullptr, Q,  S, H, H, 0);
    gemm_kernel<<<gP, 256, 0, stream>>>(hs, wk, bk, eid, nullptr, Kt, S, H, H, 0);
    gemm_kernel<<<gP, 256, 0, stream>>>(hs, wv, bv, eid, nullptr, Vt, S, H, H, 0);

    // 3) attention (scores + softmax + P@V), one block per (b, head)
    attn_kernel<<<B * NH, 128, 0, stream>>>(Q, Kt, Vt, amask, CTX);

    // 4) output projection + residual(x)
    gemm_kernel<<<gP, 256, 0, stream>>>(CTX, wo, bo, eid, hs, T1, S, H, H, 0);

    // 5) LN1
    ln_kernel<<<dim3(S, B), 256, 0, stream>>>(T1, ln1_g, ln1_b, eid, A1);

    // 6) FFN up + gelu
    dim3 gF1(FF / 128, S / 128, B);
    gemm_kernel<<<gF1, 256, 0, stream>>>(A1, w1, b1, eid, nullptr, F1, S, H, FF, 1);

    // 7) FFN down + residual(att_out)
    gemm_kernel<<<gP, 256, 0, stream>>>(F1, w2, b2, eid, A1, T2, S, FF, H, 0);

    // 8) LN2 -> final output
    ln_kernel<<<dim3(S, B), 256, 0, stream>>>(T2, ln2_g, ln2_b, eid, OUT);

    (void)in_sizes; (void)n_in; (void)out_size; (void)ws_size;
}
